// Attention_85375359910003
// MI455X (gfx1250) — matrix operand order repaired and checked by reading, not compiled, not run on hardware
//
#include <hip/hip_runtime.h>
#include <hip/hip_bf16.h>

#define BATCH 32
#define SEQ   4096
#define HDIM  512

typedef __attribute__((ext_vector_type(16))) __bf16   v16bf;
typedef __attribute__((ext_vector_type(8)))  float    v8f;
typedef __attribute__((ext_vector_type(4)))  float    v4f;
typedef __attribute__((ext_vector_type(4)))  unsigned v4u;
typedef __attribute__((ext_vector_type(8)))  unsigned v8u;

union BV { v4u q[2]; v8u u; v16bf v; };

// pack two f32's bf16-truncations into one dword: [x1.hi16 | x0.hi16]
__device__ __forceinline__ unsigned pack_hi(float x0, float x1) {
    unsigned u0 = __builtin_bit_cast(unsigned, x0);
    unsigned u1 = __builtin_bit_cast(unsigned, x1);
    return (u1 & 0xFFFF0000u) | (u0 >> 16);
}
// residual (x - trunc_bf16(x)) truncated to bf16, packed pair.
// total split error ~2^-16 relative: f32-class accuracy for the GEMM.
__device__ __forceinline__ unsigned pack_lo(float x0, float x1) {
    unsigned u0 = __builtin_bit_cast(unsigned, x0);
    unsigned u1 = __builtin_bit_cast(unsigned, x1);
    float d0 = x0 - __builtin_bit_cast(float, u0 & 0xFFFF0000u);
    float d1 = x1 - __builtin_bit_cast(float, u1 & 0xFFFF0000u);
    unsigned w0 = __builtin_bit_cast(unsigned, d0);
    unsigned w1 = __builtin_bit_cast(unsigned, d1);
    return (w1 & 0xFFFF0000u) | (w0 >> 16);
}

// branchless tanh on the TRANS pipe: 1 - 2/(2^(2*log2e*x)+1)
// x->+inf: exp->inf, rcp->0, tanh->1 ; x->-inf: exp->0, tanh->-1. No branches.
__device__ __forceinline__ float fast_tanh(float x) {
    float e = __builtin_amdgcn_exp2f(x * 2.885390082f);   // v_exp_f32
    float r = __builtin_amdgcn_rcpf(e + 1.0f);            // v_rcp_f32
    return __builtin_fmaf(-2.0f, r, 1.0f);
}

// ---------------------------------------------------------------------------
// Kernel 0: pre-swizzle Wk into bf16 hi/lo in the B-operand k order this file uses (same as A):
//   dword index = ((n_tile*16 + k_chunk)*32 + lane)*8 + p
//   lane<16: N=n_tile*16+lane,    K = k_chunk*32 + 2p (+1)
//   lane>=16: same N set,         K = k_chunk*32 + 16 + 2p (+1)
// 1 MB total, runs once, lives in L2 for every GEMM block.
// ---------------------------------------------------------------------------
__global__ void __launch_bounds__(256)
wkpack_kernel(const float* __restrict__ Wk,
              unsigned* __restrict__ wk_hi, unsigned* __restrict__ wk_lo) {
    const int L = blockIdx.x * 256 + threadIdx.x;    // 0 .. 131071 dwords
    const int p     = L & 7;
    const int lane  = (L >> 3) & 31;
    const int kc    = (L >> 8) & 15;
    const int ntile = L >> 12;
    const int hf = lane >> 4, lm = lane & 15;
    const int N  = ntile * 16 + lm;
    const int K  = kc * 32 + 16 * hf + 2 * p;
    const float x0 = Wk[(size_t)N * HDIM + K];
    const float x1 = Wk[(size_t)N * HDIM + K + 1];
    wk_hi[L] = pack_hi(x0, x1);
    wk_lo[L] = pack_lo(x0, x1);
}

// ---------------------------------------------------------------------------
// Kernel 1: q = query @ Wq^T   [B,H]  (tiny, exact f32)
// ---------------------------------------------------------------------------
__global__ void __launch_bounds__(256)
qproj_kernel(const float* __restrict__ query, const float* __restrict__ Wq,
             float* __restrict__ q) {
    const int b = blockIdx.x;
    const int t = threadIdx.x;
    const float* qr = query + (size_t)b * HDIM;
    const float* w0 = Wq + (size_t)t * HDIM;
    const float* w1 = Wq + (size_t)(t + 256) * HDIM;
    float a0 = 0.f, a1 = 0.f;
    for (int j = 0; j < HDIM; ++j) {
        float x = qr[j];
        a0 = fmaf(x, w0[j], a0);
        a1 = fmaf(x, w1[j], a1);
    }
    q[(size_t)b * HDIM + t]       = a0;
    q[(size_t)b * HDIM + t + 256] = a1;
}

// ---------------------------------------------------------------------------
// Kernel 2: fused  k = keys @ Wk^T ;  scores[b,s] = sum_h tanh(q+k)*v
// Split-bf16 WMMA. Wave owns a 16-row M tile and 8 N-tile accumulators
// (64 VGPRs) so each A-pack feeds 24 WMMAs. B comes pre-packed from L2.
// ---------------------------------------------------------------------------
__global__ void __launch_bounds__(256)
scores_kernel(const float* __restrict__ keys,
              const unsigned* __restrict__ wk_hi, const unsigned* __restrict__ wk_lo,
              const float* __restrict__ q, const float* __restrict__ v,
              float* __restrict__ scores) {
    const int b    = blockIdx.y;
    const int wave = threadIdx.x >> 5;
    const int lane = threadIdx.x & 31;
    const int hf   = lane >> 4;
    const int lm   = lane & 15;

    const int s_base = blockIdx.x * 128 + wave * 16;
    const float* krow = keys + ((size_t)b * SEQ + (s_base + lm)) * HDIM;

    const v4u* bh4 = (const v4u*)wk_hi;   // 2 x v4u per (ntile,kc,lane) slot
    const v4u* bl4 = (const v4u*)wk_lo;

    float acc[8];
#pragma unroll
    for (int j = 0; j < 8; ++j) acc[j] = 0.f;

    for (int g = 0; g < 4; ++g) {                 // N-groups of 8 tiles
        v8f C[8];
#pragma unroll
        for (int nt = 0; nt < 8; ++nt) C[nt] = v8f{};

        for (int kc = 0; kc < 16; ++kc) {         // K chunks of 32
            // ---- pack A (keys rows) once per chunk: 16 f32 -> hi/lo bf16
            const int k0 = kc * 32;
            const float* ka = krow + k0 + 16 * hf;
            v4f A0 = *(const v4f*)(ka);
            v4f A1 = *(const v4f*)(ka + 4);
            v4f A2 = *(const v4f*)(ka + 8);
            v4f A3 = *(const v4f*)(ka + 12);
            BV ah, al;
#pragma unroll
            for (int i = 0; i < 2; ++i) {
                ah.u[i]     = pack_hi(A0[2*i], A0[2*i+1]);
                ah.u[2 + i] = pack_hi(A1[2*i], A1[2*i+1]);
                ah.u[4 + i] = pack_hi(A2[2*i], A2[2*i+1]);
                ah.u[6 + i] = pack_hi(A3[2*i], A3[2*i+1]);
                al.u[i]     = pack_lo(A0[2*i], A0[2*i+1]);
                al.u[2 + i] = pack_lo(A1[2*i], A1[2*i+1]);
                al.u[4 + i] = pack_lo(A2[2*i], A2[2*i+1]);
                al.u[6 + i] = pack_lo(A3[2*i], A3[2*i+1]);
            }
#pragma unroll
            for (int nt = 0; nt < 8; ++nt) {
                const int slot = (((g * 8 + nt) * 16 + kc) * 32 + lane) * 2;
                BV bh, bl;
                bh.q[0] = bh4[slot]; bh.q[1] = bh4[slot + 1];
                bl.q[0] = bl4[slot]; bl.q[1] = bl4[slot + 1];
                C[nt] = __builtin_amdgcn_wmma_f32_16x16x32_bf16(
                            false, ah.v, false, bh.v, (short)0, C[nt], false, false);
                C[nt] = __builtin_amdgcn_wmma_f32_16x16x32_bf16(
                            false, ah.v, false, bl.v, (short)0, C[nt], false, false);
                C[nt] = __builtin_amdgcn_wmma_f32_16x16x32_bf16(
                            false, al.v, false, bh.v, (short)0, C[nt], false, false);
            }
        }
        // ---- fused epilogue: C[nt] elem j = k[M=j+8*hf, N=g*128+nt*16+lm]
#pragma unroll
        for (int nt = 0; nt < 8; ++nt) {
            const int n = g * 128 + nt * 16 + lm;
            const float qv = q[(size_t)b * HDIM + n];
            const float vv = v[n];
#pragma unroll
            for (int j = 0; j < 8; ++j)
                acc[j] = fmaf(fast_tanh(qv + C[nt][j]), vv, acc[j]);
        }
    }

    // row-sum across the 16 lanes of each half
#pragma unroll
    for (int j = 0; j < 8; ++j) {
        float t = acc[j];
        t += __shfl_xor(t, 1, 32);
        t += __shfl_xor(t, 2, 32);
        t += __shfl_xor(t, 4, 32);
        t += __shfl_xor(t, 8, 32);
        acc[j] = t;
    }
    if (lm == 0) {
        const int srow = s_base + 8 * hf;
#pragma unroll
        for (int j = 0; j < 8; ++j)
            scores[(size_t)b * SEQ + srow + j] = acc[j];
    }
}

// ---------------------------------------------------------------------------
// Kernel 3: masked softmax over S; attn -> d_out
// ---------------------------------------------------------------------------
__global__ void __launch_bounds__(256)
softmax_kernel(const float* __restrict__ scores,
               const unsigned char* __restrict__ mask,
               float* __restrict__ attn) {
    const int b    = blockIdx.x;
    const int t    = threadIdx.x;
    const int lane = t & 31;
    const int wave = t >> 5;
    __shared__ float red[8];

    float m = -2e9f;
    for (int s = t; s < SEQ; s += 256) {
        float sc = mask[(size_t)b * SEQ + s] ? scores[(size_t)b * SEQ + s] : -1e9f;
        m = fmaxf(m, sc);
    }
    for (int off = 16; off; off >>= 1) m = fmaxf(m, __shfl_xor(m, off, 32));
    if (lane == 0) red[wave] = m;
    __syncthreads();
    float bmax = -2e9f;
#pragma unroll
    for (int i = 0; i < 8; ++i) bmax = fmaxf(bmax, red[i]);
    __syncthreads();

    float sum = 0.f;
    for (int s = t; s < SEQ; s += 256) {
        float sc = mask[(size_t)b * SEQ + s] ? scores[(size_t)b * SEQ + s] : -1e9f;
        float e  = __builtin_amdgcn_exp2f((sc - bmax) * 1.442695041f);
        attn[(size_t)b * SEQ + s] = e;
        sum += e;
    }
    for (int off = 16; off; off >>= 1) sum += __shfl_xor(sum, off, 32);
    if (lane == 0) red[wave] = sum;
    __syncthreads();
    float bsum = 0.f;
#pragma unroll
    for (int i = 0; i < 8; ++i) bsum += red[i];
    const float inv = 1.0f / bsum;

    for (int s = t; s < SEQ; s += 256)
        attn[(size_t)b * SEQ + s] *= inv;
}

// ---------------------------------------------------------------------------
// Kernel 4/5: ctx = attn @ keys, deterministic two-stage reduction
// ---------------------------------------------------------------------------
__global__ void __launch_bounds__(256)
ctx_partial_kernel(const float* __restrict__ keys, const float* __restrict__ attn,
                   float* __restrict__ partial) {
    const int chunk = blockIdx.x;
    const int b     = blockIdx.y;
    const int t     = threadIdx.x;
    const float* kb = keys + ((size_t)b * SEQ + (size_t)chunk * 256) * HDIM;
    const float* ab = attn + (size_t)b * SEQ + (size_t)chunk * 256;
    float a0 = 0.f, a1 = 0.f;
    for (int s = 0; s < 256; ++s) {
        float a = ab[s];
        a0 = fmaf(a, kb[(size_t)s * HDIM + t],       a0);
        a1 = fmaf(a, kb[(size_t)s * HDIM + t + 256], a1);
    }
    float* p = partial + ((size_t)b * 16 + chunk) * HDIM;
    p[t]       = a0;
    p[t + 256] = a1;
}

__global__ void __launch_bounds__(256)
ctx_reduce_kernel(const float* __restrict__ partial, float* __restrict__ ctx) {
    const int b = blockIdx.x;
    const int t = threadIdx.x;
    float a0 = 0.f, a1 = 0.f;
#pragma unroll
    for (int c = 0; c < 16; ++c) {
        const float* p = partial + ((size_t)b * 16 + c) * HDIM;
        a0 += p[t];
        a1 += p[t + 256];
    }
    ctx[(size_t)b * HDIM + t]       = a0;
    ctx[(size_t)b * HDIM + t + 256] = a1;
}

// ---------------------------------------------------------------------------
extern "C" void kernel_launch(void* const* d_in, const int* in_sizes, int n_in,
                              void* d_out, int out_size, void* d_ws, size_t ws_size,
                              hipStream_t stream) {
    const float*         query = (const float*)d_in[0];
    const float*         keys  = (const float*)d_in[1];
    const unsigned char* mask  = (const unsigned char*)d_in[2];
    const float*         Wq    = (const float*)d_in[3];
    const float*         Wk    = (const float*)d_in[4];
    const float*         v     = (const float*)d_in[5];

    float* out  = (float*)d_out;
    float* ctx  = out;                          // [B,H]
    float* attn = out + BATCH * HDIM;           // [B,S]

    float*    ws      = (float*)d_ws;           // ~2.7 MB total
    float*    q       = ws;                     // B*H
    float*    scores  = q + BATCH * HDIM;       // B*S
    float*    partial = scores + BATCH * SEQ;   // B*16*H
    unsigned* wk_hi   = (unsigned*)(partial + BATCH * 16 * HDIM); // H*H/2 dwords
    unsigned* wk_lo   = wk_hi + (HDIM * HDIM / 2);

    wkpack_kernel<<<HDIM * HDIM / 2 / 256, 256, 0, stream>>>(Wk, wk_hi, wk_lo);
    qproj_kernel<<<BATCH, 256, 0, stream>>>(query, Wq, q);
    scores_kernel<<<dim3(SEQ / 128, BATCH), 256, 0, stream>>>(keys, wk_hi, wk_lo,
                                                              q, v, scores);
    softmax_kernel<<<BATCH, 256, 0, stream>>>(scores, mask, attn);
    ctx_partial_kernel<<<dim3(16, BATCH), 256, 0, stream>>>(keys, attn, partial);
    ctx_reduce_kernel<<<BATCH, 256, 0, stream>>>(partial, ctx);
}
